// BinaryConv2d_21466246545985
// MI455X (gfx1250) — matrix operand order repaired and checked by reading, not compiled, not run on hardware
//
#include <hip/hip_runtime.h>

typedef __attribute__((ext_vector_type(16))) _Float16 v16h;
typedef __attribute__((ext_vector_type(8)))  _Float16 v8h;
typedef __attribute__((ext_vector_type(8)))  float    v8f;
typedef __attribute__((ext_vector_type(4)))  float    f32x4;

#define IMG_H 224
#define IMG_W 224
#define CIN   64
#define COUT  64
#define NKB   18                          // K = 576 = 18 * 32
#define WPACK_HALFS (NKB * 4 * 32 * 16)   // 36864 halfs = 73728 bytes

#define XPAD    226                       // x = -1 .. 224 (zero-padded halo)
#define CSTRIDE 72                        // 64 ch + 8 pad halfs (bank decorrelation)
#define STAGE_HALFS (3 * XPAD * CSTRIDE)  // 48816 halfs = 97632 bytes LDS

using pkh = decltype(__builtin_amdgcn_cvt_pkrtz(0.0f, 0.0f));

// ---------------------------------------------------------------------------
// Prep kernel: binarize W (HWIO fp32) and pack into WMMA B-fragment order.
// Fragment (kb, g): 32 lanes x 16 f16; lane l holds column N = l%16,
// K = kb*32 + ((l>>4)&1)*16 + i, a contiguous k order this code also uses for A
// (lanes 0-15: K=0..15, lanes 16-31: K=16..31).
// ---------------------------------------------------------------------------
__global__ void pack_weights_kernel(const float* __restrict__ Wsrc,
                                    _Float16* __restrict__ Wpack) {
    int t = blockIdx.x * blockDim.x + threadIdx.x;   // 0 .. NKB*4*32-1
    if (t >= NKB * 4 * 32) return;
    int lane = t & 31;
    int g    = (t >> 5) & 3;
    int kb   = t >> 7;
    int oc    = g * 16 + (lane & 15);
    int kbase = kb * 32 + ((lane >> 4) & 1) * 16;
    _Float16* dst = Wpack + t * 16;
#pragma unroll
    for (int i = 0; i < 16; ++i) {
        float w = Wsrc[(kbase + i) * COUT + oc];
        float s = (w > 0.0f) ? 1.0f : ((w < 0.0f) ? -1.0f : 0.0f);
        dst[i] = (_Float16)s;
    }
}

// ---------------------------------------------------------------------------
// Main kernel: one block (224 thr = 7 wave32) per output row.
//  Phase 1: stage 3 padded input rows into LDS as f16 (validity resolved here).
//  Phase 2: 3x3 tap loop, A from LDS (ds pipe), B from global (vmem pipe,
//           L2/WGP$-resident), 16 v_wmma_f32_16x16x32_f16 per tap per wave.
// ---------------------------------------------------------------------------
__global__ __launch_bounds__(224) void bconv_wmma_kernel(
        const float*    __restrict__ x,
        const _Float16* __restrict__ Wpack,
        const float*    __restrict__ bias,
        float*          __restrict__ out) {
    extern __shared__ _Float16 lds[];   // STAGE_HALFS halfs

    const int tid = threadIdx.x;
    const int n   = blockIdx.x / IMG_H;   // image
    const int oy  = blockIdx.x % IMG_H;   // output row

    // ---- Phase 1: stage rows oy-1..oy+1, x=-1..224, fp32 -> f16, zero pad ----
    // One unit = 8 channels (16 B in LDS, 32 B from global).
    for (int i = tid; i < 3 * XPAD * 8; i += 224) {
        const int c8 = i & 7;            // channel group (8 ch)
        const int xr = i >> 3;
        const int xx = xr % XPAD;        // padded x: 0..225
        const int r  = xr / XPAD;        // row 0..2
        const int iy = oy + r - 1;
        const int gx = xx - 1;

        v8h val = {};
        if (iy >= 0 && iy < IMG_H && gx >= 0 && gx < IMG_W) {
            const float* p =
                x + (((long)n * IMG_H + iy) * IMG_W + gx) * CIN + c8 * 8;
            f32x4 f0 = *(const f32x4*)(p + 0);
            f32x4 f1 = *(const f32x4*)(p + 4);
            union { pkh h2[4]; v8h v; } u;
            u.h2[0] = __builtin_amdgcn_cvt_pkrtz(f0.x, f0.y);
            u.h2[1] = __builtin_amdgcn_cvt_pkrtz(f0.z, f0.w);
            u.h2[2] = __builtin_amdgcn_cvt_pkrtz(f1.x, f1.y);
            u.h2[3] = __builtin_amdgcn_cvt_pkrtz(f1.z, f1.w);
            val = u.v;
        }
        *(v8h*)&lds[(r * XPAD + xx) * CSTRIDE + c8 * 8] = val;
    }
    __syncthreads();

    // ---- Phase 2: WMMA main loop ----
    const int lane     = tid & 31;
    const int wv       = tid >> 5;          // wave 0..6 -> pixels wv*32..+31
    const int laneHalf = lane >> 4;         // 0: lanes 0-15, 1: lanes 16-31
    const int l15      = lane & 15;
    const int oxl      = wv * 32 + l15;     // subtile s adds s*16

    v8f acc[2][4] = {};

#pragma unroll
    for (int ky = 0; ky < 3; ++ky) {
#pragma unroll
        for (int kx = 0; kx < 3; ++kx) {
            const int kb0 = (ky * 3 + kx) * 2;

            // A fragments: 2 pixel subtiles x 2 channel halves, from LDS.
            // Lane-half klocal run is [+0..15] (contiguous) -> two 16B loads.
            v16h a[2][2];
#pragma unroll
            for (int s = 0; s < 2; ++s) {
                const int lx   = oxl + s * 16 + kx;          // padded x index
                const int base = (ky * XPAD + lx) * CSTRIDE + laneHalf * 16;
#pragma unroll
                for (int h = 0; h < 2; ++h) {
                    union { v8h r[2]; v16h v; } ua;
                    ua.r[0] = *(const v8h*)&lds[base + h * 32];
                    ua.r[1] = *(const v8h*)&lds[base + h * 32 + 8];
                    a[s][h] = ua.v;
                }
            }

            // B fragments from global (packed, L2/WGP$-resident); each feeds
            // both pixel subtiles.
#pragma unroll
            for (int g = 0; g < 4; ++g) {
                const v16h b0 = *(const v16h*)
                    &Wpack[(((kb0 + 0) * 4 + g) * 32 + lane) * 16];
                const v16h b1 = *(const v16h*)
                    &Wpack[(((kb0 + 1) * 4 + g) * 32 + lane) * 16];
                acc[0][g] = __builtin_amdgcn_wmma_f32_16x16x32_f16(
                    false, a[0][0], false, b0, (short)0, acc[0][g], false, false);
                acc[1][g] = __builtin_amdgcn_wmma_f32_16x16x32_f16(
                    false, a[1][0], false, b0, (short)0, acc[1][g], false, false);
                acc[0][g] = __builtin_amdgcn_wmma_f32_16x16x32_f16(
                    false, a[0][1], false, b1, (short)0, acc[0][g], false, false);
                acc[1][g] = __builtin_amdgcn_wmma_f32_16x16x32_f16(
                    false, a[1][1], false, b1, (short)0, acc[1][g], false, false);
            }
        }
    }

    // ---- Epilogue: C/D layout (reg r, lane l) -> (M = r + 8*laneHalf, N = l15)
    const long prow = ((long)n * IMG_H + oy) * IMG_W;
#pragma unroll
    for (int g = 0; g < 4; ++g) {
        const int   oc = g * 16 + l15;
        const float bv = bias[oc];
#pragma unroll
        for (int s = 0; s < 2; ++s) {
#pragma unroll
            for (int r = 0; r < 8; ++r) {
                const long p = prow + wv * 32 + s * 16 + r + laneHalf * 8;
                float v = acc[s][g][r] + bv;
                out[p * COUT + oc] = v > 0.0f ? v : 0.0f;
            }
        }
    }
}

// ---------------------------------------------------------------------------
extern "C" void kernel_launch(void* const* d_in, const int* in_sizes, int n_in,
                              void* d_out, int out_size, void* d_ws, size_t ws_size,
                              hipStream_t stream) {
    const float* x    = (const float*)d_in[0];   // 16x224x224x64 fp32 NHWC
    const float* Wsrc = (const float*)d_in[1];   // 3x3x64x64 fp32 HWIO
    const float* bias = (const float*)d_in[2];   // 64 fp32
    float*       out  = (float*)d_out;
    _Float16*    Wpack = (_Float16*)d_ws;        // 73728 B of scratch

    // 1) binarize + pack weights into B-fragment order
    const int nfrag_threads = NKB * 4 * 32;      // 2304
    pack_weights_kernel<<<(nfrag_threads + 255) / 256, 256, 0, stream>>>(Wsrc, Wpack);

    // 2) implicit-GEMM conv: one block per output row (16*224 = 3584 blocks)
    const int nblk = 16 * IMG_H;
    const size_t lds_bytes = STAGE_HALFS * sizeof(_Float16);  // 97632
    bconv_wmma_kernel<<<nblk, 224, lds_bytes, stream>>>(x, Wpack, bias, out);
}
